// DeConv2d_65506841198673
// MI455X (gfx1250) — compile-verified, matrix operand order checked by reading, not run on hardware
//
#include <hip/hip_runtime.h>

typedef __attribute__((ext_vector_type(16))) _Float16 v16h;
typedef __attribute__((ext_vector_type(8)))  _Float16 v8h;
typedef __attribute__((ext_vector_type(8)))  float    v8f;

#define NB  32
#define IC  256
#define IH  64
#define IW  64
#define OC  256
#define WPK_BYTES (OC * IC * 4 * 2)   // 256*256*4 f16 = 524288 bytes

union V16U { v16h v; v8h h[2]; };

// ---------------------------------------------------------------------------
// Repack weights (oC, iC, 2, 2) f32 -> f16 in WMMA A-fragment layout:
//   Wpk[kl][c8][o][i]  with c8 = c/8, i = c%8   (8 halves = 16B contiguous)
// Lane L of an o-tile then fetches its v16h A fragment with two b128 loads.
// ---------------------------------------------------------------------------
__global__ void pack_weights(const float* __restrict__ W, _Float16* __restrict__ Wpk) {
    int t  = blockIdx.x * blockDim.x + threadIdx.x;   // 8192 threads
    int o  = t >> 5;
    int c8 = t & 31;
    const float* src = W + (long)o * (IC * 4) + c8 * 32;  // W[o][c8*8..+8][k][l]
    float vals[8][4];
#pragma unroll
    for (int i = 0; i < 8; ++i) {
        float4 f = *(const float4*)(src + i * 4);
        vals[i][0] = f.x; vals[i][1] = f.y; vals[i][2] = f.z; vals[i][3] = f.w;
    }
#pragma unroll
    for (int kl = 0; kl < 4; ++kl) {
        v8h hv;
#pragma unroll
        for (int i = 0; i < 8; ++i) hv[i] = (_Float16)vals[i][kl];
        *(v8h*)(Wpk + ((long)(kl * 32 + c8) * 256 + o) * 8) = hv;
    }
}

// bias_sum[o][kl] = sum_c biases[o][c][k][l]
__global__ void bias_reduce(const float* __restrict__ B, float* __restrict__ bs) {
    int t = blockIdx.x * blockDim.x + threadIdx.x;    // 1024 threads
    int o = t >> 2, kl = t & 3;
    float s = 0.f;
    for (int c = 0; c < IC; ++c) s += B[((long)o * IC + c) * 4 + kl];
    bs[o * 4 + kl] = s;
}

// ---------------------------------------------------------------------------
// Main GEMM + pixel-shuffle.
// Block (256 thr = 8 waves) owns one (n,h) row: 64 pixels x 128 oC x 4 kl.
// Wave owns a 16-oC stripe; acc = v8f[kl=4][pixtile=4].
// ---------------------------------------------------------------------------
__global__ __launch_bounds__(256) void deconv_wmma(
        const float* __restrict__ X, const _Float16* __restrict__ Wpk,
        const float* __restrict__ bs, float* __restrict__ out) {
    // B-fragment staging: [c8_local(4)][px(64)][8 halves] = 4 KB
    __shared__ __align__(16) _Float16 ldsB[4 * 64 * 8];

    int wg    = blockIdx.x;
    int oHalf = wg & 1;
    int row   = wg >> 1;
    int nIdx  = row >> 6;        // 0..31
    int h     = row & 63;        // 0..63
    int tid   = threadIdx.x;
    int lane  = tid & 31;
    int wave  = tid >> 5;
    int ln    = lane & 15;
    int laneHalf = lane >> 4;    // 0: K {0-7,16-23}, 1: K {8-15,24-31}
    int oB    = oHalf * 128 + wave * 16;

    const float* xrow = X + ((long)(nIdx * IC) * IH + h) * IW;

    // cooperative X loader assignment: 32 channels x 64 w per K-step
    int cc = tid >> 3;           // 0..31 channel within chunk
    int w0 = (tid & 7) * 8;      // 8 consecutive w per thread

    v8f acc[4][4] = {};          // [kl][pixel tile]

    for (int kc = 0; kc < 8; ++kc) {     // K loop: 8 chunks of 32 channels
        int c0 = kc * 32;
        __syncthreads();                 // previous compute done reading LDS
        // stage X chunk f32 -> f16 into B-fragment layout
        const float* xp = xrow + (long)(c0 + cc) * (IH * IW) + w0;
        float4 f0 = *(const float4*)(xp);
        float4 f1 = *(const float4*)(xp + 4);
        _Float16 hv[8] = {(_Float16)f0.x, (_Float16)f0.y, (_Float16)f0.z, (_Float16)f0.w,
                          (_Float16)f1.x, (_Float16)f1.y, (_Float16)f1.z, (_Float16)f1.w};
        int lbase = ((cc >> 3) * 64) * 8 + (cc & 7);
#pragma unroll
        for (int j = 0; j < 8; ++j)
            ldsB[lbase + (w0 + j) * 8] = hv[j];
        __syncthreads();

        if (kc < 7)   // pull next chunk toward the caches (global_prefetch_b8)
            __builtin_prefetch(xrow + (long)(c0 + 32 + cc) * (IH * IW) + w0, 0, 1);

        // A fragments: one per kl, straight from L2-hot packed weights
        V16U A[4];
#pragma unroll
        for (int kl = 0; kl < 4; ++kl) {
            const _Float16* ap =
                Wpk + ((long)((kl * 32 + kc * 4 + laneHalf) * 256) + (oB + ln)) * 8;
            A[kl].h[0] = *(const v8h*)ap;                 // c8 = 4kc + half
            A[kl].h[1] = *(const v8h*)(ap + 2 * 256 * 8); // c8 = 4kc + half + 2
        }
        // 4 pixel tiles x 4 kl = 16 WMMAs (B fragment reused across kl)
#pragma unroll
        for (int t = 0; t < 4; ++t) {
            V16U Bv;
            Bv.h[0] = *(const v8h*)&ldsB[(laneHalf * 64 + t * 16 + ln) * 8];
            Bv.h[1] = *(const v8h*)&ldsB[((laneHalf + 2) * 64 + t * 16 + ln) * 8];
#pragma unroll
            for (int kl = 0; kl < 4; ++kl) {
                acc[kl][t] = __builtin_amdgcn_wmma_f32_16x16x32_f16(
                    false, A[kl].v, false, Bv.v, (short)0, acc[kl][t], false, false);
            }
        }
    }

    // bias + pixel-shuffle store: lane holds l=0 & l=1 for same w -> float2 store
    int oRow = oB + laneHalf * 8;       // lane's 8 output channels: oRow+v
    float4 bv[8];
#pragma unroll
    for (int v = 0; v < 8; ++v) bv[v] = *(const float4*)(bs + (oRow + v) * 4);

#pragma unroll
    for (int t = 0; t < 4; ++t) {
        int w = t * 16 + ln;
#pragma unroll
        for (int v = 0; v < 8; ++v) {
            int o = oRow + v;
            long rowbase = (long)(nIdx * OC + o) * 128;
#pragma unroll
            for (int k = 0; k < 2; ++k) {
                float2 val;
                float b0 = (k == 0) ? bv[v].x : bv[v].z;
                float b1 = (k == 0) ? bv[v].y : bv[v].w;
                val.x = acc[k * 2 + 0][t][v] + b0;   // l = 0
                val.y = acc[k * 2 + 1][t][v] + b1;   // l = 1
                *(float2*)(out + (rowbase + (2 * h + k)) * 128 + 2 * w) = val;
            }
        }
    }
}

extern "C" void kernel_launch(void* const* d_in, const int* in_sizes, int n_in,
                              void* d_out, int out_size, void* d_ws, size_t ws_size,
                              hipStream_t stream) {
    (void)in_sizes; (void)n_in; (void)out_size; (void)ws_size;
    const float* X = (const float*)d_in[0];   // (32,256,64,64)
    const float* W = (const float*)d_in[1];   // (256,256,2,2)
    const float* B = (const float*)d_in[2];   // (256,256,2,2)
    float* out = (float*)d_out;               // (32,256,128,128)

    _Float16* Wpk = (_Float16*)d_ws;                       // 524288 B
    float*    bsum = (float*)((char*)d_ws + WPK_BYTES);    // 4096 B

    pack_weights<<<32, 256, 0, stream>>>(W, Wpk);
    bias_reduce<<<4, 256, 0, stream>>>(B, bsum);
    deconv_wmma<<<4096, 256, 0, stream>>>(X, Wpk, bsum, out);
}
